// InverseMultiHeadAttention_5334349381727
// MI455X (gfx1250) — matrix operand order repaired and checked by reading, not compiled, not run on hardware
//
#include <hip/hip_runtime.h>

typedef __attribute__((ext_vector_type(16))) _Float16 v16h;
typedef __attribute__((ext_vector_type(8)))  float    v8f;
typedef __attribute__((ext_vector_type(4)))  int      v4i;

#define WMMA_F16(a, b, c) \
  __builtin_amdgcn_wmma_f32_16x16x32_f16(false, (a), false, (b), (short)0, (c), false, false)

// ---- gfx1250 async global->LDS copy, guarded so the file always compiles ----
#if defined(__has_builtin)
#  if __has_builtin(__builtin_amdgcn_global_load_async_to_lds_b128)
#    define ASYNC_MODE 1
#  elif __has_builtin(__builtin_amdgcn_load_to_lds)
#    define ASYNC_MODE 2
#  else
#    define ASYNC_MODE 0
#  endif
#  if __has_builtin(__builtin_amdgcn_s_wait_asynccnt)
#    define HAVE_WAIT_ASYNC_BUILTIN 1
#  else
#    define HAVE_WAIT_ASYNC_BUILTIN 0
#  endif
#else
#  define ASYNC_MODE 0
#  define HAVE_WAIT_ASYNC_BUILTIN 0
#endif

typedef __attribute__((address_space(1))) v4i*  gv4i_p;  // global (AS1) int4*
typedef __attribute__((address_space(3))) v4i*  lv4i_p;  // LDS (AS3) int4*
typedef __attribute__((address_space(3))) void* lvoid_p;

// Copy 16 bytes global -> LDS. Async when the toolchain exposes the gfx1250
// path (ASYNCcnt-tracked, no VGPR round trip); else plain b128 load/store.
__device__ __forceinline__ void copy16_g2lds(const void* g, void* l) {
#if ASYNC_MODE == 1
  // signature (probe-confirmed): (AS1 v4i*, AS3 v4i*, imm offset, imm cpol)
  __builtin_amdgcn_global_load_async_to_lds_b128(
      (gv4i_p)const_cast<void*>(g), (lv4i_p)l, 0, 0);
#elif ASYNC_MODE == 2
  __builtin_amdgcn_load_to_lds(const_cast<void*>(g), (lvoid_p)l, 16, 0, 0);
#else
  *(uint4*)l = *(const uint4*)g;
#endif
}

__device__ __forceinline__ void wait_async_copies() {
#if ASYNC_MODE == 1
#  if HAVE_WAIT_ASYNC_BUILTIN
  __builtin_amdgcn_s_wait_asynccnt(0);
#  else
  asm volatile("s_wait_asynccnt 0x0" ::: "memory");
#  endif
#endif
  // ASYNC_MODE 2 (load_to_lds) is loadcnt-tracked; compiler handles it.
}

// ---------------------------------------------------------------------------
// Fragment loaders (wave32, 16x16x32 f16 WMMA; contiguous k = 16*(L/16) + i for A and B alike)
// ---------------------------------------------------------------------------

// A matrix 16x32 (MxK): lane L -> M = L%16, K base = (L/16)*16; elements 0..15
// hold K = base..base+15.
__device__ __forceinline__ v16h frag_A(const _Float16* base, int ld, int k0) {
  const int lane = threadIdx.x & 31;
  const int m    = lane & 15;
  const int kb   = k0 + ((lane >> 4) << 4);
  const _Float16* p = base + m * ld + kb;
  v16h r;
#pragma unroll
  for (int i = 0; i < 16; ++i) r[i] = p[i];
  return r;
}

// B matrix 32x16 (KxN), source stored row-per-N (ld elems per N row),
// contraction index contiguous in memory. lane L -> N = L%16, K base = (L/16)*16.
__device__ __forceinline__ v16h frag_B_nmajor(const _Float16* base, int ld, int k0) {
  const int lane = threadIdx.x & 31;
  const int n    = lane & 15;
  const int kb   = k0 + ((lane >> 4) << 4);
  const _Float16* p = base + n * ld + kb;
  v16h r;
#pragma unroll
  for (int i = 0; i < 16; ++i) r[i] = p[i];
  return r;
}

// B matrix 32x16 (KxN), source stored row-per-K (ld elems per K row), N contiguous.
__device__ __forceinline__ v16h frag_B_kmajor(const _Float16* base, int ld, int k0) {
  const int lane = threadIdx.x & 31;
  const int n    = lane & 15;
  const int kb   = k0 + ((lane >> 4) << 4);
  v16h r;
#pragma unroll
  for (int i = 0; i < 16; ++i) r[i] = base[(kb + i) * ld + n];
  return r;
}

// ---------------------------------------------------------------------------
// GEMM: C = A @ W.T   (A: [4096,K], W: [1024,K] row-major "torch Linear" weight)
// MODE 0: A = f32, C = f16 written in head-major [B,H,S,d] layout, times scale.
// MODE 1: A = f16, C = f32 flat [4096,1024].
// Block: 256 thr (8 waves). Tile 128x128, K stepped by 32 through LDS (->f16).
// Wave w: rows (w>>2)*64, cols (w&3)*32  => 4x2 grid of 16x16 WMMA tiles.
// (Staging converts f32->f16 in flight, so it cannot use raw async DMA.)
// ---------------------------------------------------------------------------
template <int MODE>
__global__ __launch_bounds__(256) void gemm_xwT(const void* __restrict__ Ain,
                                                const float* __restrict__ W,
                                                void* __restrict__ Cout,
                                                int K, float scale) {
  constexpr int LDT = 40;  // padded LDS stride (f16 elems)
  __shared__ _Float16 sA[128 * LDT];
  __shared__ _Float16 sB[128 * LDT];

  const int tid  = threadIdx.x;
  const int wave = tid >> 5;
  const int wm   = wave >> 2;  // 0..1
  const int wn   = wave & 3;   // 0..3
  const int m0   = blockIdx.y * 128;
  const int n0   = blockIdx.x * 128;

  v8f acc[4][2];
#pragma unroll
  for (int i = 0; i < 4; ++i)
#pragma unroll
    for (int j = 0; j < 2; ++j) acc[i][j] = v8f{};

  for (int k0 = 0; k0 < K; k0 += 32) {
    // stage A and W tiles, converting to f16
#pragma unroll
    for (int i = 0; i < 16; ++i) {
      const int idx = tid + i * 256;   // 0..4095
      const int r   = idx >> 5;
      const int c   = idx & 31;
      float av;
      if (MODE == 0)
        av = ((const float*)Ain)[(size_t)(m0 + r) * K + k0 + c];
      else
        av = (float)((const _Float16*)Ain)[(size_t)(m0 + r) * K + k0 + c];
      sA[r * LDT + c] = (_Float16)av;
      sB[r * LDT + c] = (_Float16)W[(size_t)(n0 + r) * K + k0 + c];
    }
    if (k0 + 32 < K) {
      __builtin_prefetch(W + (size_t)n0 * K + k0 + 32 + tid * 32, 0, 1);
    }
    __syncthreads();

    v16h bf[2];
#pragma unroll
    for (int j = 0; j < 2; ++j)
      bf[j] = frag_B_nmajor(sB + (wn * 32 + j * 16) * LDT, LDT, 0);
#pragma unroll
    for (int i = 0; i < 4; ++i) {
      v16h af = frag_A(sA + (wm * 64 + i * 16) * LDT, LDT, 0);
#pragma unroll
      for (int j = 0; j < 2; ++j) acc[i][j] = WMMA_F16(af, bf[j], acc[i][j]);
    }
    __syncthreads();
  }

  // C layout: VGPR r, lane L -> M = r + 8*(L/16), N = L%16
  const int lane = tid & 31;
  const int nl   = lane & 15;
  const int hi   = lane >> 4;
#pragma unroll
  for (int i = 0; i < 4; ++i)
#pragma unroll
    for (int j = 0; j < 2; ++j)
#pragma unroll
      for (int r = 0; r < 8; ++r) {
        const int gm = m0 + wm * 64 + i * 16 + r + 8 * hi;
        const int gn = n0 + wn * 32 + j * 16 + nl;
        const float v = acc[i][j][r] * scale;
        if (MODE == 0) {
          const int b = gm >> 11, s = gm & 2047;  // S = 2048
          const int h = gn >> 7, dim = gn & 127;  // d = 128
          ((_Float16*)Cout)[(((size_t)(b * 8 + h) * 2048) + s) * 128 + dim] =
              (_Float16)v;
        } else {
          ((float*)Cout)[(size_t)gm * 1024 + gn] = v;
        }
      }
}

// ---------------------------------------------------------------------------
// Per-head V column sums: vsum[bh][d] = sum_s V[bh][s][d]
// (att = 1 - softmax  =>  out = colsum(V) - softmax(S) @ V)
// ---------------------------------------------------------------------------
__global__ __launch_bounds__(128) void vsum_kernel(const _Float16* __restrict__ vh,
                                                   float* __restrict__ vsum) {
  const int bh = blockIdx.x;
  const int d  = threadIdx.x;
  float s = 0.f;
  for (int t = 0; t < 2048; ++t)
    s += (float)vh[((size_t)bh * 2048 + t) * 128 + d];
  vsum[bh * 128 + d] = s;
}

// ---------------------------------------------------------------------------
// Flash attention with inverse-softmax epilogue.
// grid = (S/128, B*H); block = 256 thr (8 waves). Wave owns 16 q rows; Q frags
// pinned in VGPRs. K/V double-buffered in dynamic LDS: tile t+1's async DMA
// (gfx1250 ASYNCcnt path) overlaps tile t's WMMA work. One barrier per tile:
// each wave waits its own asynccnt then barriers, making all DMA writes
// visible and proving every wave is done reading the buffer being restaged.
// ---------------------------------------------------------------------------
constexpr int LKV   = 136;          // padded stride for 128-d rows (16B mult)
constexpr int LP    = 72;           // padded stride for 64-key P rows
constexpr int KVBUF = 64 * LKV;     // one K or V tile, f16 elems (8704)
constexpr int FLASH_LDS_BYTES = (4 * KVBUF + 8 * 16 * LP) * 2;  // 88064 B

__global__ __launch_bounds__(256) void flash_inv_kernel(
    const _Float16* __restrict__ qh, const _Float16* __restrict__ kh,
    const _Float16* __restrict__ vh, const float* __restrict__ vsum,
    _Float16* __restrict__ attout) {
  extern __shared__ _Float16 smem[];
  // layout: [ K buf0 | K buf1 | V buf0 | V buf1 | per-wave P ]
  // (pointers computed with runtime arithmetic only — a brace-initialized
  //  pointer array here becomes a static addrspacecast initializer that
  //  ld.lld rejects)

  const int tid  = threadIdx.x;
  const int lane = tid & 31;
  const int wave = tid >> 5;
  const int bh   = blockIdx.y;
  const int q0   = blockIdx.x * 128 + wave * 16;
  const size_t headbase = (size_t)bh * 2048 * 128;

  _Float16* pw = smem + 4 * KVBUF + wave * (16 * LP);

  // stage one 64x128 K tile + V tile into buffer `buf` (async DMA, no wait)
  auto stage = [&](int kt, int buf) {
    const _Float16* gK = kh + headbase + (size_t)kt * 64 * 128;
    const _Float16* gV = vh + headbase + (size_t)kt * 64 * 128;
    _Float16* dK = smem + buf * KVBUF;
    _Float16* dV = smem + 2 * KVBUF + buf * KVBUF;
#pragma unroll
    for (int i = 0; i < 4; ++i) {
      const int idx = tid + i * 256;  // 0..1023
      const int r   = idx >> 4;
      const int c8  = idx & 15;
      copy16_g2lds(gK + r * 128 + c8 * 8, dK + r * LKV + c8 * 8);
      copy16_g2lds(gV + r * 128 + c8 * 8, dV + r * LKV + c8 * 8);
    }
  };

  // Q fragments (scale already folded in by projection): 16 rows x 128 dims
  v16h aq[4];
#pragma unroll
  for (int c = 0; c < 4; ++c)
    aq[c] = frag_A(qh + headbase + (size_t)q0 * 128, 128, c * 32);

  v8f o[8];
#pragma unroll
  for (int i = 0; i < 8; ++i) o[i] = v8f{};
  float mrow[8], lrow[8];
#pragma unroll
  for (int r = 0; r < 8; ++r) { mrow[r] = -1e30f; lrow[r] = 0.f; }

  stage(0, 0);  // prologue

  for (int kt = 0; kt < 32; ++kt) {
    const int buf = kt & 1;
    wait_async_copies();   // current tile's DMA (issued last iteration) done
    __syncthreads();       // all waves' writes visible; old buffer fully read
    if (kt + 1 < 32) stage(kt + 1, buf ^ 1);  // overlap next DMA with compute

    const _Float16* cK = smem + buf * KVBUF;
    const _Float16* cV = smem + 2 * KVBUF + buf * KVBUF;

    // scores: 16 x 64 = 4 n-tiles, contraction over d=128 (4 chunks of 32)
    v8f sc[4];
#pragma unroll
    for (int nt = 0; nt < 4; ++nt) {
      v8f a = v8f{};
#pragma unroll
      for (int c = 0; c < 4; ++c) {
        v16h bf = frag_B_nmajor(cK + (nt * 16) * LKV, LKV, c * 32);
        a = WMMA_F16(aq[c], bf, a);
      }
      sc[nt] = a;
    }

    // online softmax: row reductions across 16-lane halves
    float mn[8], al[8];
#pragma unroll
    for (int r = 0; r < 8; ++r) {
      float mx = sc[0][r];
#pragma unroll
      for (int nt = 1; nt < 4; ++nt) mx = fmaxf(mx, sc[nt][r]);
#pragma unroll
      for (int off = 1; off < 16; off <<= 1)
        mx = fmaxf(mx, __shfl_xor(mx, off, 32));
      mn[r]   = fmaxf(mrow[r], mx);
      al[r]   = __expf(mrow[r] - mn[r]);
      mrow[r] = mn[r];
    }
    float rs[8];
#pragma unroll
    for (int r = 0; r < 8; ++r) rs[r] = 0.f;
#pragma unroll
    for (int nt = 0; nt < 4; ++nt)
#pragma unroll
      for (int r = 0; r < 8; ++r) {
        const float p = __expf(sc[nt][r] - mn[r]);
        sc[nt][r] = p;
        rs[r] += p;
      }
#pragma unroll
    for (int r = 0; r < 8; ++r) {
#pragma unroll
      for (int off = 1; off < 16; off <<= 1) rs[r] += __shfl_xor(rs[r], off, 32);
      lrow[r] = lrow[r] * al[r] + rs[r];
    }
#pragma unroll
    for (int i = 0; i < 8; ++i)
#pragma unroll
      for (int r = 0; r < 8; ++r) o[i][r] *= al[r];

    // P: C-layout -> A-layout through per-wave LDS (wave-private, no barrier)
    {
      const int nl = lane & 15, hi = lane >> 4;
#pragma unroll
      for (int nt = 0; nt < 4; ++nt)
#pragma unroll
        for (int r = 0; r < 8; ++r)
          pw[(r + 8 * hi) * LP + nt * 16 + nl] = (_Float16)sc[nt][r];
    }

    // O += P(16x64) @ V(64x128): 2 key chunks x 8 dim tiles
#pragma unroll
    for (int c = 0; c < 2; ++c) {
      v16h ap = frag_A(pw, LP, c * 32);
#pragma unroll
      for (int nt = 0; nt < 8; ++nt) {
        v16h bv = frag_B_kmajor(cV + nt * 16, LKV, c * 32);
        o[nt] = WMMA_F16(ap, bv, o[nt]);
      }
    }
  }

  // epilogue: out = colsum(V) - o/l, written as [B, S, H*d] f16
  const int nl = lane & 15, hi = lane >> 4;
  const int b = bh >> 3, h = bh & 7;
#pragma unroll
  for (int nt = 0; nt < 8; ++nt) {
    const float vs = vsum[bh * 128 + nt * 16 + nl];
#pragma unroll
    for (int r = 0; r < 8; ++r) {
      const int s    = q0 + r + 8 * hi;
      const float vo = vs - o[nt][r] / lrow[r];
      attout[((size_t)(b * 2048 + s)) * 1024 + h * 128 + nt * 16 + nl] =
          (_Float16)vo;
    }
  }
}

// ---------------------------------------------------------------------------
extern "C" void kernel_launch(void* const* d_in, const int* in_sizes, int n_in,
                              void* d_out, int out_size, void* d_ws, size_t ws_size,
                              hipStream_t stream) {
  (void)in_sizes; (void)n_in; (void)out_size; (void)ws_size;
  const float* q  = (const float*)d_in[0];
  const float* k  = (const float*)d_in[1];
  const float* v  = (const float*)d_in[2];
  const float* Wq = (const float*)d_in[3];
  const float* Wk = (const float*)d_in[4];
  const float* Wv = (const float*)d_in[5];
  const float* Wo = (const float*)d_in[6];

  const size_t BSD = (size_t)2 * 2048 * 1024;  // 4,194,304 elems
  _Float16* qh     = (_Float16*)d_ws;
  _Float16* kh     = qh + BSD;
  _Float16* vh     = kh + BSD;
  _Float16* attout = vh + BSD;
  float*    vsum   = (float*)(attout + BSD);

  const dim3 gemm_grid(1024 / 128, 4096 / 128);  // (8, 32)
  const float scale = 0.08838834764831845f;      // 128^-0.5

  gemm_xwT<0><<<gemm_grid, 256, 0, stream>>>(q, Wq, qh, 1024, scale);
  gemm_xwT<0><<<gemm_grid, 256, 0, stream>>>(k, Wk, kh, 1024, 1.0f);
  gemm_xwT<0><<<gemm_grid, 256, 0, stream>>>(v, Wv, vh, 1024, 1.0f);
  vsum_kernel<<<16, 128, 0, stream>>>(vh, vsum);
  flash_inv_kernel<<<dim3(16, 16), 256, FLASH_LDS_BYTES, stream>>>(
      qh, kh, vh, vsum, attout);
  gemm_xwT<1><<<gemm_grid, 256, 0, stream>>>(attout, Wo, d_out, 1024, 1.0f);
}
